// Classifier_5755256176863
// MI455X (gfx1250) — compile-verified, matrix operand order checked by reading, not run on hardware
//
#include <hip/hip_runtime.h>
#include <math.h>

typedef __bf16 bf16_t;
typedef __attribute__((ext_vector_type(16))) __bf16 v16bf;
typedef __attribute__((ext_vector_type(8)))  __bf16 v8bf;
typedef __attribute__((ext_vector_type(8)))  float  v8f;

#define HDIM  128
#define GCNT  128
#define POOLW 384
#define OUTC  10
#define BN_EPS 1e-5f
#define WFRAG_ELEMS (8 * 4 * 32 * 16)   // 16384 bf16 = 32 KB

// ---------------- elementwise / utility kernels ----------------

__global__ void zero_f32_kernel(float* __restrict__ p, int n) {
  int i = blockIdx.x * blockDim.x + threadIdx.x;
  if (i < n) p[i] = 0.f;
}

__global__ void copy_f32x4_kernel(const float4* __restrict__ a, float4* __restrict__ b, int n4) {
  int i = blockIdx.x * blockDim.x + threadIdx.x;
  if (i < n4) b[i] = a[i];
}

__global__ void cvt_f32_bf16_kernel(const float* __restrict__ a, bf16_t* __restrict__ b, int n) {
  int i = blockIdx.x * blockDim.x + threadIdx.x;
  if (i < n) b[i] = (bf16_t)a[i];
}

// scatter-add over edges: h[dst] += x[src]  (h pre-initialized with x)
__global__ void scatter_add_edges_kernel(const float* __restrict__ x,
                                         const int* __restrict__ src,
                                         const int* __restrict__ dst,
                                         float* __restrict__ h, int nedges) {
  int t = blockIdx.x * blockDim.x + threadIdx.x;
  int e = t >> 5;              // 32 threads per edge, 4 floats each
  if (e >= nedges) return;
  int f = (t & 31) << 2;
  int s = src[e], d = dst[e];
  float4 v = *(const float4*)(x + (size_t)s * HDIM + f);
  float* o = h + (size_t)d * HDIM + f;
  atomicAdd(o + 0, v.x);
  atomicAdd(o + 1, v.y);
  atomicAdd(o + 2, v.z);
  atomicAdd(o + 3, v.w);
}

// pack a [128x128] f32 row-major weight into bf16 WMMA B-fragment order:
// Wf[((ntile*4 + kc)*32 + lane)*16 + e], e in 0..15 maps to
// k = kc*32 + (lane&16 ? 8 : 0) + (e<8 ? e : e+8), n = ntile*16 + (lane&15)
__global__ void pack_weight_frag_kernel(const float* __restrict__ W, bf16_t* __restrict__ Wf) {
  int t = blockIdx.x * blockDim.x + threadIdx.x;   // WFRAG_ELEMS total
  if (t >= WFRAG_ELEMS) return;
  int e     = t & 15;
  int lane  = (t >> 4) & 31;
  int kc    = (t >> 9) & 3;
  int ntile = t >> 11;
  int k = kc * 32 + ((lane & 16) ? 8 : 0) + (e < 8 ? e : e + 8);
  int n = ntile * 16 + (lane & 15);
  Wf[t] = (bf16_t)W[k * HDIM + n];
}

// ---------------- WMMA GEMM: C[M,128] = A[M,128](bf16) @ W[128,128] + bias ----------------
// 8 waves/block, one wave per 16-row tile; weight fragments staged in LDS (32 KB);
// A buffer is padded to a multiple of 128 rows, so fragment loads are unguarded.
__global__ __launch_bounds__(256) void gemm_n128_wmma_kernel(
    const bf16_t* __restrict__ A, const bf16_t* __restrict__ Wfrag,
    const float* __restrict__ bias, float* __restrict__ C, int nrows, int do_relu) {
  __shared__ __align__(16) bf16_t sW[WFRAG_ELEMS];

  // cooperative stage of the packed weight into LDS: 2048 x uint4 by 256 threads
  {
    const uint4* g = (const uint4*)Wfrag;
    uint4* s = (uint4*)sW;
    const int tid = threadIdx.x;
#pragma unroll
    for (int j = 0; j < 8; ++j) s[tid + j * 256] = g[tid + j * 256];
  }
  __syncthreads();   // all waves reach this before any wave-uniform early return

  const int lane  = threadIdx.x & 31;
  const int wave  = threadIdx.x >> 5;
  const int mtile = blockIdx.x * 8 + wave;
  if (mtile * 16 >= nrows) return;                 // wave-uniform: EXEC stays all-ones

  const int arow = mtile * 16 + (lane & 15);
  const int hsel = (lane & 16) ? 8 : 0;            // K sub-offset for upper half-wave
  const bf16_t* abase = A + (size_t)arow * HDIM + hsel;

  v8f acc[8];
  v8f zacc = {0.f, 0.f, 0.f, 0.f, 0.f, 0.f, 0.f, 0.f};
#pragma unroll
  for (int t = 0; t < 8; ++t) acc[t] = zacc;

#pragma unroll
  for (int kc = 0; kc < 4; ++kc) {
    v8bf lo = *(const v8bf*)(abase + kc * 32);        // K = base .. base+7
    v8bf hi = *(const v8bf*)(abase + kc * 32 + 16);   // K = base+16 .. base+23
    v16bf a;
#pragma unroll
    for (int i = 0; i < 8; ++i) { a[i] = lo[i]; a[8 + i] = hi[i]; }
#pragma unroll
    for (int t = 0; t < 8; ++t) {
      const v16bf b = *(const v16bf*)(sW + (((t * 4 + kc) * 32 + lane) << 4));
      acc[t] = __builtin_amdgcn_wmma_f32_16x16x32_bf16(
          /*neg_a=*/false, a, /*neg_b=*/false, b,
          /*c_mod=*/(short)0, acc[t], /*reuse_a=*/false, /*reuse_b=*/false);
    }
  }

  // D layout: lane holds col = lane&15; VGPR r holds row = r + (lane&16 ? 8 : 0)
  const int rbase = mtile * 16 + hsel;
  const int col   = lane & 15;
#pragma unroll
  for (int t = 0; t < 8; ++t) {
    const int n = t * 16 + col;
    const float bn = bias[n];
#pragma unroll
    for (int r = 0; r < 8; ++r) {
      const int m = rbase + r;
      if (m < nrows) {
        float v = acc[t][r] + bn;
        if (do_relu) v = fmaxf(v, 0.f);
        C[(size_t)m * HDIM + n] = v;
      }
    }
  }
}

// ---------------- BatchNorm (training-mode batch stats) ----------------

__global__ __launch_bounds__(128) void bn_stats_kernel(const float* __restrict__ h,
                                                       float* __restrict__ sums,
                                                       float* __restrict__ sumsq, int nrows) {
  int c = threadIdx.x;                             // column 0..127
  float s = 0.f, sq = 0.f;
  for (int r = blockIdx.x; r < nrows; r += gridDim.x) {
    float v = h[(size_t)r * HDIM + c];
    s += v; sq += v * v;
  }
  atomicAdd(&sums[c], s);
  atomicAdd(&sumsq[c], sq);
}

__global__ __launch_bounds__(128) void bn_finalize_kernel(const float* __restrict__ sums,
                                                          const float* __restrict__ sumsq,
                                                          const float* __restrict__ gamma,
                                                          const float* __restrict__ beta,
                                                          float* __restrict__ scale,
                                                          float* __restrict__ shift, float invN) {
  int c = threadIdx.x;
  float mean = sums[c] * invN;
  float var  = sumsq[c] * invN - mean * mean;      // biased variance
  float s = gamma[c] * rsqrtf(var + BN_EPS);
  scale[c] = s;
  shift[c] = beta[c] - mean * s;
}

// h -> bf16(relu(h*scale + shift))
__global__ void bn_apply_relu_cvt_kernel(const float* __restrict__ h,
                                         const float* __restrict__ scale,
                                         const float* __restrict__ shift,
                                         bf16_t* __restrict__ out, int n) {
  int i = blockIdx.x * blockDim.x + threadIdx.x;
  if (i >= n) return;
  int c = i & (HDIM - 1);
  float v = fmaf(h[i], scale[c], shift[c]);
  out[i] = (bf16_t)fmaxf(v, 0.f);
}

// ---------------- graph pooling + heads ----------------

__global__ void pool_scatter_kernel(const float* __restrict__ xl, const int* __restrict__ batch,
                                    float* __restrict__ pooled, int coloff, int nrows) {
  int t = blockIdx.x * blockDim.x + threadIdx.x;
  int nidx = t >> 7;
  if (nidx >= nrows) return;
  int f = t & 127;
  atomicAdd(&pooled[(size_t)batch[nidx] * POOLW + coloff + f], xl[(size_t)nidx * HDIM + f]);
}

__global__ __launch_bounds__(64) void heads_kernel(const float* __restrict__ pooled,
                                                   const float* __restrict__ clsW,
                                                   const float* __restrict__ clsb,
                                                   const float* __restrict__ wmW1,
                                                   const float* __restrict__ wmb1,
                                                   const float* __restrict__ wmW2,
                                                   const float* __restrict__ wmb2,
                                                   float* __restrict__ out) {
  __shared__ float p[POOLW];
  __shared__ float hh[64];
  int g = blockIdx.x;
  for (int i = threadIdx.x; i < POOLW; i += 64) p[i] = pooled[(size_t)g * POOLW + i];
  __syncthreads();

  if (threadIdx.x < OUTC) {
    float s = clsb[threadIdx.x];
    for (int c = 0; c < POOLW; ++c) s = fmaf(p[c], clsW[c * OUTC + threadIdx.x], s);
    out[g * OUTC + threadIdx.x] = s;
  }
  {
    int j = threadIdx.x;
    float s = wmb1[j];
    for (int c = 0; c < POOLW; ++c) s = fmaf(p[c], wmW1[c * 64 + j], s);
    hh[j] = fmaxf(s, 0.f);
  }
  __syncthreads();
  if (threadIdx.x == 0) {
    float s = wmb2[0];
    for (int j = 0; j < 64; ++j) s = fmaf(hh[j], wmW2[j], s);
    out[GCNT * OUTC + g] = 1.f / (1.f + expf(-s));
  }
}

// ---------------- launch ----------------

extern "C" void kernel_launch(void* const* d_in, const int* in_sizes, int n_in,
                              void* d_out, int out_size, void* d_ws, size_t ws_size,
                              hipStream_t stream) {
  const int N = in_sizes[0] / HDIM;
  const int E = in_sizes[1] / 2;
  const int NH = N * HDIM;
  const int gGemm = (N + 127) / 128;
  const int Npad = gGemm * 128;                    // pad rows so A-fragment loads are unguarded

  const float* x     = (const float*)d_in[0];
  const int*   ei    = (const int*)d_in[1];
  const int*   srcv  = ei;
  const int*   dstv  = ei + E;
  const int*   batch = (const int*)d_in[2];

  const float *W1[3], *b1[3], *gam[3], *bet[3], *W2[3], *b2[3];
  for (int l = 0; l < 3; ++l) {
    int base = 3 + l * 6;
    W1[l]  = (const float*)d_in[base + 0];
    b1[l]  = (const float*)d_in[base + 1];
    gam[l] = (const float*)d_in[base + 2];
    bet[l] = (const float*)d_in[base + 3];
    W2[l]  = (const float*)d_in[base + 4];
    b2[l]  = (const float*)d_in[base + 5];
  }
  const float* clsW = (const float*)d_in[21];
  const float* clsb = (const float*)d_in[22];
  const float* wmW1 = (const float*)d_in[23];
  const float* wmb1 = (const float*)d_in[24];
  const float* wmW2 = (const float*)d_in[25];
  const float* wmb2 = (const float*)d_in[26];

  // workspace carving (256B aligned slabs)
  char* ws = (char*)d_ws;
  size_t off = 0;
  auto carve = [&](size_t bytes) -> char* {
    char* p = ws + off;
    off += (bytes + 255) & ~(size_t)255;
    return p;
  };
  float* xl[3];
  for (int l = 0; l < 3; ++l) xl[l] = (float*)carve((size_t)NH * 4);
  float*  hbuf = (float*)carve((size_t)NH * 4);
  bf16_t* hbf  = (bf16_t*)carve((size_t)Npad * HDIM * 2);   // padded for unguarded loads
  bf16_t* Wfrag[6];
  for (int i = 0; i < 6; ++i) Wfrag[i] = (bf16_t*)carve((size_t)WFRAG_ELEMS * 2);
  float* stats  = (float*)carve(4 * HDIM * 4);   // sums | sumsq | scale | shift
  float* sums   = stats;
  float* sumsq  = stats + HDIM;
  float* scale  = stats + 2 * HDIM;
  float* shift  = stats + 3 * HDIM;
  float* pooled = (float*)carve((size_t)GCNT * POOLW * 4);

  const int TB = 256;
  const int gNH  = (NH + TB - 1) / TB;
  const int gNH4 = (NH / 4 + TB - 1) / TB;
  const int gE   = (E * 32 + TB - 1) / TB;

  // pack all 6 weight matrices into WMMA fragment order (every call; deterministic)
  for (int l = 0; l < 3; ++l) {
    pack_weight_frag_kernel<<<64, TB, 0, stream>>>(W1[l], Wfrag[2 * l + 0]);
    pack_weight_frag_kernel<<<64, TB, 0, stream>>>(W2[l], Wfrag[2 * l + 1]);
  }
  // zero pooled accumulator
  zero_f32_kernel<<<(GCNT * POOLW + TB - 1) / TB, TB, 0, stream>>>(pooled, GCNT * POOLW);

  const float* cur = x;
  for (int l = 0; l < 3; ++l) {
    // h = x + sum_{j->i} x_j
    copy_f32x4_kernel<<<gNH4, TB, 0, stream>>>((const float4*)cur, (float4*)hbuf, NH / 4);
    scatter_add_edges_kernel<<<gE, TB, 0, stream>>>(cur, srcv, dstv, hbuf, E);
    cvt_f32_bf16_kernel<<<gNH, TB, 0, stream>>>(hbuf, hbf, NH);
    // lin1 = h @ W1 + b1   (reuse hbuf as output; hbf is the input)
    gemm_n128_wmma_kernel<<<gGemm, TB, 0, stream>>>(hbf, Wfrag[2 * l + 0], b1[l], hbuf, N, 0);
    // batchnorm stats + apply + relu + cvt
    zero_f32_kernel<<<1, TB, 0, stream>>>(sums, 2 * HDIM);
    bn_stats_kernel<<<512, 128, 0, stream>>>(hbuf, sums, sumsq, N);
    bn_finalize_kernel<<<1, 128, 0, stream>>>(sums, sumsq, gam[l], bet[l], scale, shift, 1.f / (float)N);
    bn_apply_relu_cvt_kernel<<<gNH, TB, 0, stream>>>(hbuf, scale, shift, hbf, NH);
    // x_{l+1} = relu(relu(bn) @ W2 + b2)   (outer relu fused)
    gemm_n128_wmma_kernel<<<gGemm, TB, 0, stream>>>(hbf, Wfrag[2 * l + 1], b2[l], xl[l], N, 1);
    // pooled[:, l*128:(l+1)*128] += segment_sum(x_{l+1})
    pool_scatter_kernel<<<gNH, TB, 0, stream>>>(xl[l], batch, pooled, l * HDIM, N);
    cur = xl[l];
  }

  heads_kernel<<<GCNT, 64, 0, stream>>>(pooled, clsW, clsb, wmW1, wmb1, wmW2, wmb2, (float*)d_out);
}
